// MultiheadLatentAttentionV3_2_13262859010626
// MI455X (gfx1250) — matrix operand order repaired and checked by reading, not compiled, not run on hardware
//
#include <hip/hip_runtime.h>
#include <math.h>

// ---------------- problem constants ----------------
#define Bc   2
#define Sc   2048
#define Hc   2048
#define NHc  16
#define QLc  1536
#define KVLc 512
#define DKc  128
#define DRc  64
#define DVc  128
#define DQK  192      // DK + DR
#define IHc  8
#define IDc  128
#define TOPKc 8
#define BSc  (Bc*Sc)  // 4096

typedef unsigned short bf16_t;
typedef __attribute__((ext_vector_type(4)))  unsigned int v4u;
typedef __attribute__((ext_vector_type(8)))  unsigned int v8u;
typedef __attribute__((ext_vector_type(16))) __bf16      v16bf;
typedef __attribute__((ext_vector_type(8)))  float       v8f;

__device__ __forceinline__ bf16_t f2bf(float f) {
  union { float f; unsigned u; } v; v.f = f;
  unsigned u = v.u;
  u += 0x7fffu + ((u >> 16) & 1u);           // round-to-nearest-even
  return (bf16_t)(u >> 16);
}
__device__ __forceinline__ float bf2f(bf16_t h) {
  union { unsigned u; float f; } v; v.u = ((unsigned)h) << 16;
  return v.f;
}

// Build a 16-element bf16 fragment from two contiguous 8-element (16B) chunks.
__device__ __forceinline__ v16bf frag16(const bf16_t* p0, const bf16_t* p1) {
  const uint4 lo = *(const uint4*)p0;
  const uint4 hi = *(const uint4*)p1;
  v8u u;
  u[0] = lo.x; u[1] = lo.y; u[2] = lo.z; u[3] = lo.w;
  u[4] = hi.x; u[5] = hi.y; u[6] = hi.z; u[7] = hi.w;
  return __builtin_bit_cast(v16bf, u);
}

// ---------------- TDM 2D tile load: global(row-major) -> LDS(row-major) ------
// Descriptor per CDNA5 ISA 8.3/8.4: group0 = {count/lds_addr/global_addr/type},
// group1 = {data_size, tensor dims, tile dims, dim0 stride}. data_size = 2B.
__device__ __forceinline__ void tdm_load_2d(unsigned lds_off, const bf16_t* g,
                                            unsigned dim0, unsigned dim1,
                                            unsigned tile0, unsigned tile1,
                                            unsigned stride0) {
  unsigned long long ga = (unsigned long long)(const void*)g;
  v4u g0;
  g0[0] = 1u;                                        // count=1, user descriptor
  g0[1] = lds_off;                                   // LDS byte address
  g0[2] = (unsigned)ga;                              // global addr [31:0]
  g0[3] = (unsigned)((ga >> 32) & 0x01FFFFFFu) | (2u << 30);  // addr[56:32] | type=2
  v8u g1;
  g1[0] = 0x10000u;                                  // data_size = 1 (2 bytes)
  g1[1] = (dim0 & 0xFFFFu) << 16;                    // tensor_dim0[15:0]
  g1[2] = (dim0 >> 16) | ((dim1 & 0xFFFFu) << 16);   // dim0[31:16] | dim1[15:0]
  g1[3] = (dim1 >> 16) | (tile0 << 16);              // dim1[31:16] | tile_dim0
  g1[4] = tile1;                                     // tile_dim1 (tile_dim2=0)
  g1[5] = stride0;                                   // tensor_dim0_stride[31:0]
  g1[6] = 0u;
  g1[7] = 0u;
  asm volatile("tensor_load_to_lds %0, %1" :: "s"(g0), "s"(g1) : "memory");
}

// ---------------- fp32 -> bf16 convert ----------------
__global__ __launch_bounds__(256) void k_f2bf(const float* __restrict__ src,
                                              bf16_t* __restrict__ dst, size_t n) {
  size_t i = (size_t)blockIdx.x * blockDim.x + threadIdx.x;
  if (i < n) dst[i] = f2bf(src[i]);
}

// fp32 (K x N, row-major) -> bf16 transposed (N x K, row-major)
__global__ __launch_bounds__(256) void k_f2bfT(const float* __restrict__ src,
                                               bf16_t* __restrict__ dst,
                                               int K, int N) {
  size_t i = (size_t)blockIdx.x * blockDim.x + threadIdx.x;
  if (i >= (size_t)K * N) return;
  int k = (int)(i / N), n = (int)(i % N);
  dst[(size_t)n * K + k] = f2bf(src[i]);
}

// ---------------- WMMA bf16 GEMM: C[M,N] = A[M,K] * Wt[N,K]^T ----------------
// block = 256 threads (8 waves); block tile 128(M) x 64(N); K stepped by 32.
// TDM double-buffered staging of A (128x32) and B^T (64x32) tiles.
__global__ __launch_bounds__(256) void k_gemm_bf16(const bf16_t* __restrict__ A,
                                                   const bf16_t* __restrict__ Wt,
                                                   void* __restrict__ Cout,
                                                   int M, int N, int K, int c_is_f32) {
  __shared__ __align__(16) bf16_t As[2][128][32];
  __shared__ __align__(16) bf16_t BsT[2][64][32];
  const int bm = blockIdx.y * 128, bn = blockIdx.x * 64;
  const int tid = threadIdx.x;
  const int wave = tid >> 5, lane = tid & 31;
  const int wm = wave * 16;
  const int lx = lane & 15;
  const bool hi = (lane >= 16);

  v8f acc[4];
#pragma unroll
  for (int t = 0; t < 4; ++t) acc[t] = (v8f){};

  const int nk = K / 32;
  const unsigned ldsA0 = (unsigned)(size_t)&As[0][0][0];
  const unsigned ldsA1 = (unsigned)(size_t)&As[1][0][0];
  const unsigned ldsB0 = (unsigned)(size_t)&BsT[0][0][0];
  const unsigned ldsB1 = (unsigned)(size_t)&BsT[1][0][0];

  if (wave == 0) {  // preload tile 0
    tdm_load_2d(ldsA0, A + (size_t)bm * K, (unsigned)K, 128u, 32u, 128u, (unsigned)K);
    tdm_load_2d(ldsB0, Wt + (size_t)bn * K, (unsigned)K, 64u, 32u, 64u, (unsigned)K);
  }

  int buf = 0;
  for (int it = 0; it < nk; ++it) {
    const int k0 = it * 32;
    if (wave == 0) {
      if (it + 1 < nk) {
        const unsigned la = buf ? ldsA0 : ldsA1;
        const unsigned lb = buf ? ldsB0 : ldsB1;
        tdm_load_2d(la, A + (size_t)bm * K + k0 + 32, (unsigned)(K - k0 - 32), 128u,
                    32u, 128u, (unsigned)K);
        tdm_load_2d(lb, Wt + (size_t)bn * K + k0 + 32, (unsigned)(K - k0 - 32), 64u,
                    32u, 64u, (unsigned)K);
        __builtin_amdgcn_s_wait_tensorcnt((short)2);   // current tile done
      } else {
        __builtin_amdgcn_s_wait_tensorcnt((short)0);
      }
    }
    __syncthreads();   // tiles ready

    // A fragment: two contiguous 16B LDS chunks (ISA 16-bit A layout)
    const int kb = hi ? 8 : 0;
    const bf16_t* arow = &As[buf][wm + lx][0];
    v16bf af = frag16(arow + kb, arow + kb + 16);
    // 4 B fragments: contiguous chunks from transposed tile (ISA 16-bit B layout)
    const int kb2 = hi ? 16 : 0;
#pragma unroll
    for (int t = 0; t < 4; ++t) {
      const bf16_t* brow = &BsT[buf][t * 16 + lx][kb];
      v16bf bfv = frag16(brow, brow + 16);
      acc[t] = __builtin_amdgcn_wmma_f32_16x16x32_bf16(false, af, false, bfv,
                                                       (short)0, acc[t], false, false);
    }
    __syncthreads();   // done reading this buffer
    buf ^= 1;
  }

  // store: C layout -> row = r + 8*hi, col = lane&15 per 16-col tile
  const int rbase = bm + wm + (hi ? 8 : 0);
  if (c_is_f32) {
    float* C = (float*)Cout;
#pragma unroll
    for (int t = 0; t < 4; ++t) {
      int col = bn + t * 16 + lx;
#pragma unroll
      for (int r = 0; r < 8; ++r)
        C[(size_t)(rbase + r) * N + col] = acc[t][r];
    }
  } else {
    bf16_t* C = (bf16_t*)Cout;
#pragma unroll
    for (int t = 0; t < 4; ++t) {
      int col = bn + t * 16 + lx;
#pragma unroll
      for (int r = 0; r < 8; ++r)
        C[(size_t)(rbase + r) * N + col] = f2bf(acc[t][r]);
    }
  }
}

// ---------------- RoPE + pack q/k/v ----------------
// qout/kout: (B, NH, S, 192); vout transposed: (B, NH, DV, S)
__global__ __launch_bounds__(256) void k_pack(const bf16_t* __restrict__ qc,
                                              const bf16_t* __restrict__ qr,
                                              const bf16_t* __restrict__ kc,
                                              const bf16_t* __restrict__ krb,
                                              const bf16_t* __restrict__ vfull,
                                              bf16_t* __restrict__ qout,
                                              bf16_t* __restrict__ kout,
                                              bf16_t* __restrict__ vout) {
  const int s = blockIdx.x, b = blockIdx.y;
  const size_t bs = (size_t)b * Sc + s;

  for (int i = threadIdx.x; i < NHc * DQK; i += blockDim.x) {
    int h = i / DQK, d = i % DQK;
    bf16_t qv, kv;
    if (d < DKc) {
      qv = qc[bs * 2048 + h * DKc + d];
      kv = kc[bs * 2048 + h * DKc + d];
    } else {
      int dr = d - DKc;
      int pair = dr >> 1;
      float inv = powf(10000.f, -(float)(2 * pair) / (float)DRc);
      float fr = (float)s * inv;
      float cs = __cosf(fr), sn = __sinf(fr);
      float qxr = bf2f(qr[bs * 1024 + h * DRc + 2 * pair]);
      float qxi = bf2f(qr[bs * 1024 + h * DRc + 2 * pair + 1]);
      float kxr = bf2f(krb[bs * DRc + 2 * pair]);
      float kxi = bf2f(krb[bs * DRc + 2 * pair + 1]);
      if (dr & 1) { qv = f2bf(qxr * sn + qxi * cs); kv = f2bf(kxr * sn + kxi * cs); }
      else        { qv = f2bf(qxr * cs - qxi * sn); kv = f2bf(kxr * cs - kxi * sn); }
    }
    size_t o = (((size_t)b * NHc + h) * Sc + s) * DQK + d;
    qout[o] = qv; kout[o] = kv;
  }
  for (int i = threadIdx.x; i < NHc * DVc; i += blockDim.x) {
    int h = i / DVc, dv = i % DVc;
    vout[(((size_t)b * NHc + h) * DVc + dv) * Sc + s] = vfull[bs * 2048 + h * DVc + dv];
  }
}

// ---------------- flash attention (one wave per 16-query tile) ----------------
__global__ __launch_bounds__(32) void k_flash(const bf16_t* __restrict__ q,
                                              const bf16_t* __restrict__ k,
                                              const bf16_t* __restrict__ v,
                                              float* __restrict__ attnf) {
  const int qt = blockIdx.x, h = blockIdx.y, b = blockIdx.z;
  const int lane = threadIdx.x;
  const int lx = lane & 15;
  const bool hi = (lane >= 16);
  const int rofs = hi ? 8 : 0;

  const size_t headqk = ((size_t)b * NHc + h) * (size_t)Sc * DQK;
  const bf16_t* qh = q + headqk;
  const bf16_t* kh = k + headqk;
  const bf16_t* vh = v + ((size_t)b * NHc + h) * (size_t)DVc * Sc;  // (DV, S)

  __shared__ __align__(16) bf16_t Ps[16][32];

  // Q fragments: 6 chunks of K=32 over d=192, b128 loads
  const int kb = hi ? 8 : 0;
  v16bf qf[6];
  {
    const bf16_t* qrow = qh + (size_t)(qt * 16 + lx) * DQK;
#pragma unroll
    for (int c6 = 0; c6 < 6; ++c6)
      qf[c6] = frag16(qrow + c6 * 32 + kb, qrow + c6 * 32 + kb + 16);
  }

  const float scale = 0.07216878364870322f; // 1/sqrt(192)
  v8f m, l, o[8];
#pragma unroll
  for (int r = 0; r < 8; ++r) { m[r] = -1e30f; l[r] = 0.f; }
#pragma unroll
  for (int c = 0; c < 8; ++c) o[c] = (v8f){};

  const int kb2 = hi ? 16 : 0;
  const int nkt = qt / 2 + 1;
  for (int kt = 0; kt < nkt; ++kt) {
    const int kbase = kt * 32;
    if (kt + 1 < nkt)
      __builtin_prefetch(kh + (size_t)(kbase + 32 + lane) * DQK, 0, 1);

    v8f sarr[2]; sarr[0] = (v8f){}; sarr[1] = (v8f){};
#pragma unroll
    for (int t = 0; t < 2; ++t) {
      const bf16_t* krow = kh + (size_t)(kbase + t * 16 + lx) * DQK + kb;
#pragma unroll
      for (int c6 = 0; c6 < 6; ++c6) {
        v16bf bfv = frag16(krow + c6 * 32, krow + c6 * 32 + 16);
        sarr[t] = __builtin_amdgcn_wmma_f32_16x16x32_bf16(false, qf[c6], false, bfv,
                                                          (short)0, sarr[t], false, false);
      }
    }
    // scale + causal mask
    const int col0 = kbase + lx, col1 = col0 + 16;
#pragma unroll
    for (int r = 0; r < 8; ++r) {
      int qi = qt * 16 + r + rofs;
      sarr[0][r] = (col0 <= qi) ? sarr[0][r] * scale : -1e30f;
      sarr[1][r] = (col1 <= qi) ? sarr[1][r] * scale : -1e30f;
    }
    // online softmax: row reductions across the 16 column lanes
    v8f mt;
#pragma unroll
    for (int r = 0; r < 8; ++r) mt[r] = fmaxf(sarr[0][r], sarr[1][r]);
#pragma unroll
    for (int off = 1; off < 16; off <<= 1) {
#pragma unroll
      for (int r = 0; r < 8; ++r) mt[r] = fmaxf(mt[r], __shfl_xor(mt[r], off, 32));
    }
    v8f alpha;
#pragma unroll
    for (int r = 0; r < 8; ++r) {
      float mn = fmaxf(m[r], mt[r]);
      alpha[r] = __expf(m[r] - mn);
      m[r] = mn;
    }
#pragma unroll
    for (int r = 0; r < 8; ++r) {
      sarr[0][r] = __expf(sarr[0][r] - m[r]);
      sarr[1][r] = __expf(sarr[1][r] - m[r]);
    }
    v8f lt;
#pragma unroll
    for (int r = 0; r < 8; ++r) lt[r] = sarr[0][r] + sarr[1][r];
#pragma unroll
    for (int off = 1; off < 16; off <<= 1) {
#pragma unroll
      for (int r = 0; r < 8; ++r) lt[r] += __shfl_xor(lt[r], off, 32);
    }
#pragma unroll
    for (int r = 0; r < 8; ++r) l[r] = l[r] * alpha[r] + lt[r];
#pragma unroll
    for (int c = 0; c < 8; ++c)
#pragma unroll
      for (int r = 0; r < 8; ++r) o[c][r] *= alpha[r];

    // stage P (16x32) via LDS: C-matrix layout -> A-fragment layout
    __syncthreads();
#pragma unroll
    for (int r = 0; r < 8; ++r) {
      Ps[r + rofs][lx]      = f2bf(sarr[0][r]);
      Ps[r + rofs][lx + 16] = f2bf(sarr[1][r]);
    }
    __syncthreads();
    v16bf pf = frag16(&Ps[lx][kb], &Ps[lx][kb + 16]);

    // o += P(16x32) @ V(32x128): V transposed (DV,S) -> contiguous b128 chunks
#pragma unroll
    for (int c = 0; c < 8; ++c) {
      const bf16_t* vrow = vh + (size_t)(c * 16 + lx) * Sc + kbase + kb;
      v16bf vf = frag16(vrow, vrow + 16);
      o[c] = __builtin_amdgcn_wmma_f32_16x16x32_bf16(false, pf, false, vf,
                                                     (short)0, o[c], false, false);
    }
  }

  // epilogue: normalize, store to (B, S, NH*DV) fp32
#pragma unroll
  for (int c = 0; c < 8; ++c) {
    int dv = c * 16 + lx;
#pragma unroll
    for (int r = 0; r < 8; ++r) {
      int si = qt * 16 + r + rofs;
      attnf[(((size_t)b * Sc + si) * NHc + h) * DVc + dv] = o[c][r] / l[r];
    }
  }
}

// ---------------- sparse indexer branch (tiny, scalar fp32) ----------------
__global__ __launch_bounds__(256) void k_gate(const float* __restrict__ x,
                                              const float* __restrict__ Wg,
                                              float* __restrict__ gate) {
  int i = blockIdx.x * blockDim.x + threadIdx.x;
  if (i >= BSc * IHc) return;
  int ih = i % IHc;
  size_t bs = i / IHc;
  float acc = 0.f;
  const float* xr = x + bs * Hc;
  for (int j = 0; j < Hc; ++j) acc += xr[j] * Wg[(size_t)j * IHc + ih];
  gate[bs * IHc + ih] = acc;
}

__global__ void k_topk(const float* __restrict__ gate, int* __restrict__ topidx) {
  int p = blockIdx.x;                  // (b*IH + ih)
  int b = p / IHc, ih = p % IHc;
  int chosen[TOPKc];
  for (int t = 0; t < TOPKc; ++t) {
    float best = -1e38f; int bi = 0;
    for (int s = 0; s < Sc; ++s) {
      bool used = false;
      for (int u = 0; u < t; ++u) used |= (chosen[u] == s);
      if (used) continue;
      float gv = gate[((size_t)b * Sc + s) * IHc + ih];
      if (gv > best) { best = gv; bi = s; }
    }
    chosen[t] = bi;
    topidx[p * TOPKc + t] = bi;
  }
}

__global__ __launch_bounds__(128) void k_sel(const float* __restrict__ x,
                                             const float* __restrict__ Widx,
                                             const int* __restrict__ topidx,
                                             float* __restrict__ sel) {
  int i = blockIdx.x * blockDim.x + threadIdx.x;
  if (i >= Bc * IHc * TOPKc * IDc) return;
  int d = i % IDc;
  int t = (i / IDc) % TOPKc;
  int ih = (i / (IDc * TOPKc)) % IHc;
  int b = i / (IDc * TOPKc * IHc);
  int s = topidx[(b * IHc + ih) * TOPKc + t];
  const float* xr = x + ((size_t)b * Sc + s) * Hc;
  float acc = 0.f;
  for (int j = 0; j < Hc; ++j) acc += xr[j] * Widx[(size_t)j * (IHc * IDc) + ih * IDc + d];
  sel[i] = acc;
}

__global__ __launch_bounds__(128) void k_sqkv(const float* __restrict__ sel,
                                              const float* __restrict__ Wsq,
                                              const float* __restrict__ Wsk,
                                              const float* __restrict__ Wsv,
                                              float* __restrict__ sq,
                                              float* __restrict__ sk,
                                              float* __restrict__ sv) {
  const int NSEL = Bc * IHc * TOPKc * IDc;
  int i = blockIdx.x * blockDim.x + threadIdx.x;
  if (i >= 3 * NSEL) return;
  int which = i / NSEL;
  int r = i % NSEL;
  int d = r % IDc;
  int row = r / IDc;                   // (b,ih,t)
  const float* W = which == 0 ? Wsq : (which == 1 ? Wsk : Wsv);
  float acc = 0.f;
  const float* sr = sel + (size_t)row * IDc;
  for (int j = 0; j < IDc; ++j) acc += sr[j] * W[(size_t)j * IDc + d];
  (which == 0 ? sq : (which == 1 ? sk : sv))[r] = acc;
}

__global__ __launch_bounds__(128) void k_sattn(const float* __restrict__ sq,
                                               const float* __restrict__ sk,
                                               const float* __restrict__ sv,
                                               float* __restrict__ sout) {
  int i = blockIdx.x * blockDim.x + threadIdx.x;
  if (i >= Bc * IHc * TOPKc * IDc) return;
  int d = i % IDc;
  int t = (i / IDc) % TOPKc;
  int bh = i / (IDc * TOPKc);          // (b*IH + ih)
  const float inv = 0.08838834764831845f;   // 1/sqrt(128)
  float sc[TOPKc];
  float mx = -1e30f;
  for (int kk = 0; kk < TOPKc; ++kk) {
    float a = 0.f;
    const float* qrow = sq + ((size_t)bh * TOPKc + t) * IDc;
    const float* krow = sk + ((size_t)bh * TOPKc + kk) * IDc;
    for (int j = 0; j < IDc; ++j) a += qrow[j] * krow[j];
    sc[kk] = a * inv;
    mx = fmaxf(mx, sc[kk]);
  }
  float den = 0.f;
  for (int kk = 0; kk < TOPKc; ++kk) { sc[kk] = __expf(sc[kk] - mx); den += sc[kk]; }
  float acc = 0.f;
  for (int kk = 0; kk < TOPKc; ++kk)
    acc += sc[kk] * sv[((size_t)bh * TOPKc + kk) * IDc + d];
  sout[i] = acc / den;
}

__global__ __launch_bounds__(128) void k_iodv(const float* __restrict__ sout,
                                              const float* __restrict__ Wiout,
                                              float* __restrict__ iodv) {
  int i = blockIdx.x * blockDim.x + threadIdx.x;
  if (i >= Bc * IHc * DVc) return;
  int dv = i % DVc;
  int bh = i / DVc;                    // (b*IH + ih)
  float acc = 0.f;
  for (int t = 0; t < TOPKc; ++t) {
    const float* srow = sout + ((size_t)bh * TOPKc + t) * IDc;
    for (int j = 0; j < IDc; ++j) acc += srow[j] * Wiout[(size_t)j * Hc + dv];
  }
  iodv[i] = acc * (1.0f / TOPKc);
}

// ---------------- add io correction + convert attn to bf16 ----------------
__global__ __launch_bounds__(256) void k_addcvt(const float* __restrict__ attnf,
                                                const float* __restrict__ iodv,
                                                bf16_t* __restrict__ attnb) {
  size_t i = (size_t)blockIdx.x * blockDim.x + threadIdx.x;
  if (i >= (size_t)BSc * (NHc * DVc)) return;
  int dv = i % DVc;
  int h = (i / DVc) % NHc;
  size_t bs = i / (DVc * NHc);
  int b = (int)(bs / Sc);
  float add = iodv[((size_t)b * IHc + (h >> 1)) * DVc + dv];
  attnb[i] = f2bf(attnf[i] + add);
}

// ---------------- host side ----------------
static size_t align_up(size_t v, size_t a) { return (v + a - 1) & ~(a - 1); }

extern "C" void kernel_launch(void* const* d_in, const int* in_sizes, int n_in,
                              void* d_out, int out_size, void* d_ws, size_t ws_size,
                              hipStream_t stream) {
  (void)in_sizes; (void)n_in; (void)out_size; (void)ws_size;
  const float* x     = (const float*)d_in[0];
  const float* W_c   = (const float*)d_in[1];
  const float* W_cp  = (const float*)d_in[2];
  const float* W_qc  = (const float*)d_in[3];
  const float* W_qr  = (const float*)d_in[4];
  const float* W_kc  = (const float*)d_in[5];
  const float* W_kr  = (const float*)d_in[6];
  const float* W_v   = (const float*)d_in[7];
  const float* W_o   = (const float*)d_in[8];
  const float* W_idx = (const float*)d_in[9];
  const float* W_gate= (const float*)d_in[10];
  const float* W_sq  = (const float*)d_in[11];
  const float* W_sk  = (const float*)d_in[12];
  const float* W_sv  = (const float*)d_in[13];
  const float* W_iout= (const float*)d_in[14];

  char* base = (char*)d_ws;
  size_t off = 0;
  auto alloc = [&](size_t bytes) { char* p = base + off; off = align_up(off + bytes, 256); return (void*)p; };

  bf16_t* xb    = (bf16_t*)alloc((size_t)BSc * Hc * 2);
  bf16_t* Wc_t  = (bf16_t*)alloc((size_t)Hc * KVLc * 2);      // (N,K) transposed
  bf16_t* Wcp_t = (bf16_t*)alloc((size_t)Hc * QLc * 2);
  bf16_t* Wqc_t = (bf16_t*)alloc((size_t)QLc * 2048 * 2);
  bf16_t* Wqr_t = (bf16_t*)alloc((size_t)QLc * 1024 * 2);
  bf16_t* Wkc_t = (bf16_t*)alloc((size_t)KVLc * 2048 * 2);
  bf16_t* Wkr_t = (bf16_t*)alloc((size_t)Hc * DRc * 2);
  bf16_t* Wv_t  = (bf16_t*)alloc((size_t)KVLc * 2048 * 2);
  bf16_t* Wo_t  = (bf16_t*)alloc((size_t)2048 * 2048 * 2);
  bf16_t* cbf   = (bf16_t*)alloc((size_t)BSc * KVLc * 2);
  bf16_t* cpbf  = (bf16_t*)alloc((size_t)BSc * QLc * 2);
  bf16_t* qc    = (bf16_t*)alloc((size_t)BSc * 2048 * 2);
  bf16_t* qr    = (bf16_t*)alloc((size_t)BSc * 1024 * 2);
  bf16_t* kc    = (bf16_t*)alloc((size_t)BSc * 2048 * 2);
  bf16_t* krb   = (bf16_t*)alloc((size_t)BSc * DRc * 2);
  bf16_t* vfull = (bf16_t*)alloc((size_t)BSc * 2048 * 2);
  bf16_t* qB    = (bf16_t*)alloc((size_t)Bc * NHc * Sc * DQK * 2);
  bf16_t* kB    = (bf16_t*)alloc((size_t)Bc * NHc * Sc * DQK * 2);
  bf16_t* vB    = (bf16_t*)alloc((size_t)Bc * NHc * Sc * DVc * 2);
  float*  attnf = (float*)alloc((size_t)BSc * 2048 * 4);
  bf16_t* attnb = (bf16_t*)alloc((size_t)BSc * 2048 * 2);
  float*  gate  = (float*)alloc((size_t)BSc * IHc * 4);
  int*    topidx= (int*)alloc((size_t)Bc * IHc * TOPKc * 4);
  float*  sel   = (float*)alloc((size_t)Bc * IHc * TOPKc * IDc * 4);
  float*  sq    = (float*)alloc((size_t)Bc * IHc * TOPKc * IDc * 4);
  float*  sk    = (float*)alloc((size_t)Bc * IHc * TOPKc * IDc * 4);
  float*  sv    = (float*)alloc((size_t)Bc * IHc * TOPKc * IDc * 4);
  float*  sout  = (float*)alloc((size_t)Bc * IHc * TOPKc * IDc * 4);
  float*  iodv  = (float*)alloc((size_t)Bc * IHc * DVc * 4);

  // x -> bf16 (row-major); weights -> bf16 transposed (N,K)
  {
    size_t n = (size_t)BSc * Hc;
    k_f2bf<<<dim3((unsigned)((n + 255) / 256)), dim3(256), 0, stream>>>(x, xb, n);
  }
  auto convT = [&](const float* s, bf16_t* d, int K, int N) {
    size_t n = (size_t)K * N;
    k_f2bfT<<<dim3((unsigned)((n + 255) / 256)), dim3(256), 0, stream>>>(s, d, K, N);
  };
  convT(W_c,  Wc_t,  Hc,  KVLc);
  convT(W_cp, Wcp_t, Hc,  QLc);
  convT(W_qc, Wqc_t, QLc, 2048);
  convT(W_qr, Wqr_t, QLc, 1024);
  convT(W_kc, Wkc_t, KVLc,2048);
  convT(W_kr, Wkr_t, Hc,  DRc);
  convT(W_v,  Wv_t,  KVLc,2048);
  convT(W_o,  Wo_t,  2048,2048);

  auto gemm = [&](const bf16_t* A, const bf16_t* Wt, void* C, int M, int N, int K, int f32out) {
    k_gemm_bf16<<<dim3(N / 64, M / 128), dim3(256), 0, stream>>>(A, Wt, C, M, N, K, f32out);
  };
  gemm(xb,   Wc_t,  cbf,   BSc, KVLc, Hc,  0);   // c
  gemm(xb,   Wcp_t, cpbf,  BSc, QLc,  Hc,  0);   // cp
  gemm(cpbf, Wqc_t, qc,    BSc, 2048, QLc, 0);   // q_c
  gemm(cpbf, Wqr_t, qr,    BSc, 1024, QLc, 0);   // q_r
  gemm(cbf,  Wkc_t, kc,    BSc, 2048, KVLc,0);   // k_c
  gemm(xb,   Wkr_t, krb,   BSc, DRc,  Hc,  0);   // k_r base
  gemm(cbf,  Wv_t,  vfull, BSc, 2048, KVLc,0);   // v

  k_pack<<<dim3(Sc, Bc), dim3(256), 0, stream>>>(qc, qr, kc, krb, vfull, qB, kB, vB);

  k_flash<<<dim3(Sc / 16, NHc, Bc), dim3(32), 0, stream>>>(qB, kB, vB, attnf);

  k_gate<<<dim3((BSc * IHc + 255) / 256), dim3(256), 0, stream>>>(x, W_gate, gate);
  k_topk<<<dim3(Bc * IHc), dim3(1), 0, stream>>>(gate, topidx);
  k_sel<<<dim3((Bc * IHc * TOPKc * IDc + 127) / 128), dim3(128), 0, stream>>>(x, W_idx, topidx, sel);
  k_sqkv<<<dim3((3 * Bc * IHc * TOPKc * IDc + 127) / 128), dim3(128), 0, stream>>>(
      sel, W_sq, W_sk, W_sv, sq, sk, sv);
  k_sattn<<<dim3((Bc * IHc * TOPKc * IDc + 127) / 128), dim3(128), 0, stream>>>(sq, sk, sv, sout);
  k_iodv<<<dim3((Bc * IHc * DVc + 127) / 128), dim3(128), 0, stream>>>(sout, W_iout, iodv);

  k_addcvt<<<dim3((unsigned)(((size_t)BSc * 2048 + 255) / 256)), dim3(256), 0, stream>>>(
      attnf, iodv, attnb);

  gemm(attnb, Wo_t, d_out, BSc, Hc, 2048, 1);
}
